// CustomEncoderLayer_23751169147345
// MI455X (gfx1250) — matrix operand order repaired and checked by reading, not compiled, not run on hardware
//
#include <hip/hip_runtime.h>

#define BB 8
#define SS 1024
#define DDIM 512
#define HH 8
#define HDIM 64
#define FFD 128
#define NTOK (BB * SS)

typedef __attribute__((ext_vector_type(16))) __bf16 v16bf;
typedef __attribute__((ext_vector_type(8)))  float  v8f;

union BF16x16 { v16bf v; unsigned int u[8]; };

__device__ __forceinline__ unsigned short f2bf(float f) {
    unsigned int u = __float_as_uint(f);
    u += 0x7FFFu + ((u >> 16) & 1u);   // round-to-nearest-even
    return (unsigned short)(u >> 16);
}

__device__ __forceinline__ v8f wmma_bf16(v16bf a, v16bf b, v8f c) {
    return __builtin_amdgcn_wmma_f32_16x16x32_bf16(
        /*neg_a=*/false, a, /*neg_b=*/false, b,
        /*c_mod=*/(short)0, c, /*reuse_a=*/false, /*reuse_b=*/false);
}

// A-tile loader: 16x32 bf16 tile from row-major M[ld] at (m0,k0).
// ISA layout: lane m=lane&15, half=lane>>4; reg j holds K pair:
//   j<4 : K = half*8 + 2j ; j>=4 : K = 16 + half*8 + 2(j-4)
__device__ __forceinline__ v16bf load_a_tile(const unsigned short* p, int ld, int m0, int k0) {
    int lane = threadIdx.x & 31;
    int m = lane & 15, hlf = lane >> 4;
    BF16x16 r;
    const unsigned short* row = p + (size_t)(m0 + m) * ld + k0;
#pragma unroll
    for (int j = 0; j < 8; ++j) {
        int kk = (j < 4) ? (hlf * 8 + 2 * j) : (16 + hlf * 8 + 2 * (j - 4));
        r.u[j] = *(const unsigned int*)(row + kk);
    }
    return r.v;
}

// B-tile loader from row-major B^T[N][K] at (n0,k0) -> 32x16 B operand.
// ISA layout: lane n=lane&15, half=lane>>4; reg j holds K pair:
//   j<4 : K = half*8 + 2j ; j>=4 : K = 16 + half*8 + 2(j-4)
__device__ __forceinline__ v16bf load_bt_tile(const unsigned short* p, int ld, int n0, int k0) {
    int lane = threadIdx.x & 31;
    int n = lane & 15, hlf = lane >> 4;
    BF16x16 r;
    const unsigned short* row = p + (size_t)(n0 + n) * ld + k0;
#pragma unroll
    for (int j = 0; j < 8; ++j) r.u[j] = *(const unsigned int*)(row + ((j < 4) ? (hlf * 8 + 2 * j) : (16 + hlf * 8 + 2 * (j - 4))));
    return r.v;
}

// A-operand read from an LDS-staged 16x32 bf16 tile (ds path).
__device__ __forceinline__ v16bf load_a_lds(const unsigned short* ab) {
    int lane = threadIdx.x & 31;
    int m = lane & 15, hlf = lane >> 4;
    BF16x16 r;
#pragma unroll
    for (int j = 0; j < 8; ++j) {
        int kk = (j < 4) ? (hlf * 8 + 2 * j) : (16 + hlf * 8 + 2 * (j - 4));
        r.u[j] = *(const unsigned int*)(ab + m * 32 + kk);
    }
    return r.v;
}

// CDNA5 async DMA: global -> LDS, 8 bytes per lane, tracked by ASYNCcnt.
__device__ __forceinline__ void async_copy_b64(void* ldsDst, const void* g) {
    unsigned ldsOff = (unsigned)(unsigned long long)ldsDst;  // low 32 bits = LDS byte offset
    asm volatile("global_load_async_to_lds_b64 %0, %1, off"
                 :: "v"(ldsOff), "v"(g) : "memory");
}

// ---- 16-lane butterfly reductions via DPP16 (no LDS, no DScnt waits) ----
// steps: quad_perm xor1 (0xB1), quad_perm xor2 (0x4E),
//        ROW_HALF_MIRROR (0x141), ROW_MIRROR (0x140)
__device__ __forceinline__ float rmax16(float v) {
    v = fmaxf(v, __int_as_float(__builtin_amdgcn_update_dpp(0, __float_as_int(v), 0xB1,  0xF, 0xF, true)));
    v = fmaxf(v, __int_as_float(__builtin_amdgcn_update_dpp(0, __float_as_int(v), 0x4E,  0xF, 0xF, true)));
    v = fmaxf(v, __int_as_float(__builtin_amdgcn_update_dpp(0, __float_as_int(v), 0x141, 0xF, 0xF, true)));
    v = fmaxf(v, __int_as_float(__builtin_amdgcn_update_dpp(0, __float_as_int(v), 0x140, 0xF, 0xF, true)));
    return v;
}
__device__ __forceinline__ float rsum16(float v) {
    v += __int_as_float(__builtin_amdgcn_update_dpp(0, __float_as_int(v), 0xB1,  0xF, 0xF, true));
    v += __int_as_float(__builtin_amdgcn_update_dpp(0, __float_as_int(v), 0x4E,  0xF, 0xF, true));
    v += __int_as_float(__builtin_amdgcn_update_dpp(0, __float_as_int(v), 0x141, 0xF, 0xF, true));
    v += __int_as_float(__builtin_amdgcn_update_dpp(0, __float_as_int(v), 0x140, 0xF, 0xF, true));
    return v;
}

// ---------------- elementwise f32 -> bf16 convert ----------------
__global__ void k_f32_to_bf16(const float* __restrict__ in, unsigned short* __restrict__ out, int n) {
    int i = blockIdx.x * blockDim.x + threadIdx.x;
    if (i < n) out[i] = f2bf(in[i]);
}

// =====================================================================
// Async-staged GEMM core: 4 waves/WG share a double-buffered LDS A tile
// (16 x 32 bf16, filled via global_load_async_to_lds_b64); each wave owns
// TN 16x16 accumulators (16 x TN*16 output).  B read direct from global
// as row-major B^T[N][K].
// =====================================================================
template <int TN, int MODE>   // MODE 0 = bias/resid/relu epilogue, 1 = QKV scatter
__global__ void k_gemm_async(const unsigned short* __restrict__ A,
                             const unsigned short* __restrict__ BT,
                             const float* __restrict__ bias,
                             const float* __restrict__ resid,
                             float* __restrict__ outF,
                             unsigned short* __restrict__ outB,
                             unsigned short* __restrict__ qO,
                             unsigned short* __restrict__ kO,
                             unsigned short* __restrict__ vTO,
                             int N, int K, int relu) {
    __shared__ __align__(16) unsigned short aBuf[2][16 * 32];
    const int tid  = threadIdx.x;         // 0..127
    const int wave = tid >> 5;
    const int lane = tid & 31;
    const int n    = lane & 15, hlf = lane >> 4;
    const int m0   = blockIdx.y * 16;
    const int n0   = blockIdx.x * (4 * TN * 16) + wave * (TN * 16);

    v8f acc[TN];
#pragma unroll
    for (int t = 0; t < TN; ++t) acc[t] = (v8f){};

    const int row = tid >> 3;             // 0..15 : tile row this thread stages
    const int cu  = (tid & 7) * 4;        // ushort offset in row (8-byte chunks)
    const int ksteps = K >> 5;

    // prologue: stage k-step 0 into buffer 0
    async_copy_b64(&aBuf[0][row * 32 + cu], A + (size_t)(m0 + row) * K + cu);

    for (int i = 0; i < ksteps; ++i) {
        if (i + 1 < ksteps) {
            async_copy_b64(&aBuf[(i + 1) & 1][row * 32 + cu],
                           A + (size_t)(m0 + row) * K + ((i + 1) << 5) + cu);
            asm volatile("s_wait_asynccnt 0x1" ::: "memory");  // buffer i complete
        } else {
            asm volatile("s_wait_asynccnt 0x0" ::: "memory");
        }
        __syncthreads();                                       // all 4 waves' stages visible
        v16bf av = load_a_lds(&aBuf[i & 1][0]);
        int k0 = i << 5;
#pragma unroll
        for (int t = 0; t < TN; ++t)
            acc[t] = wmma_bf16(av, load_bt_tile(BT, K, n0 + t * 16, k0), acc[t]);
        __syncthreads();   // reads done before this buffer is re-staged (next iter)
    }

    if (MODE == 0) {
#pragma unroll
        for (int t = 0; t < TN; ++t) {
            int nc = n0 + t * 16 + n;
            float bv = bias ? bias[nc] : 0.f;
#pragma unroll
            for (int r = 0; r < 8; ++r) {
                int m = m0 + r + 8 * hlf;
                size_t idx = (size_t)m * N + nc;
                float v = acc[t][r] + bv;
                if (resid) v += resid[idx];
                if (relu)  v = fmaxf(v, 0.f);
                if (outF) outF[idx] = v;
                if (outB) outB[idx] = f2bf(v);
            }
        }
    } else {
        // QKV scatter: q,k -> [B,H,S,64] ; v -> transposed [B,H,64,S]
#pragma unroll
        for (int t = 0; t < TN; ++t) {
            int ng = n0 + t * 16 + n;
            float bv = bias[ng];
            int which = ng / DDIM;        // uniform within a 16-wide tile
            int d = ng % DDIM;
            int h = d / HDIM, dh = d % HDIM;
#pragma unroll
            for (int r = 0; r < 8; ++r) {
                int m = m0 + r + 8 * hlf;
                int b = m / SS, s = m % SS;
                unsigned short val = f2bf(acc[t][r] + bv);
                size_t bh = (size_t)(b * HH + h);
                if (which == 0)      qO[(bh * SS + s) * HDIM + dh] = val;
                else if (which == 1) kO[(bh * SS + s) * HDIM + dh] = val;
                else                 vTO[(bh * HDIM + dh) * SS + s] = val;
            }
        }
    }
}

// ---------------- flash-style attention ----------------
// one wave handles 16 query rows of one (b,h); online softmax over 32-key blocks
__global__ void k_attn(const unsigned short* __restrict__ qB,
                       const unsigned short* __restrict__ kB,
                       const unsigned short* __restrict__ vT,
                       unsigned short* __restrict__ ctxB) {
    __shared__ __align__(16) unsigned short ldsP[16 * 32]; // P tile: C-layout -> A-layout
    const int lane = threadIdx.x & 31;
    const int n = lane & 15, hlf = lane >> 4;
    const int qt = blockIdx.x, h = blockIdx.y, b = blockIdx.z;

    const unsigned short* qp = qB + ((size_t)(b * HH + h) * SS) * HDIM;
    const unsigned short* kp = kB + ((size_t)(b * HH + h) * SS) * HDIM;
    const unsigned short* vp = vT + ((size_t)(b * HH + h) * HDIM) * SS;

    v16bf qa0 = load_a_tile(qp, HDIM, qt * 16, 0);
    v16bf qa1 = load_a_tile(qp, HDIM, qt * 16, 32);

    v8f o0 = {}, o1 = {}, o2 = {}, o3 = {};
    float mst[8], lst[8];
#pragma unroll
    for (int r = 0; r < 8; ++r) { mst[r] = -1e30f; lst[r] = 0.f; }

    for (int key0 = 0; key0 < SS; key0 += 32) {
        v8f s0 = {}, s1 = {};
        s0 = wmma_bf16(qa0, load_bt_tile(kp, HDIM, key0, 0), s0);
        s0 = wmma_bf16(qa1, load_bt_tile(kp, HDIM, key0, 32), s0);
        s1 = wmma_bf16(qa0, load_bt_tile(kp, HDIM, key0 + 16, 0), s1);
        s1 = wmma_bf16(qa1, load_bt_tile(kp, HDIM, key0 + 16, 32), s1);

#pragma unroll
        for (int r = 0; r < 8; ++r) {
            float a = s0[r] * 0.125f;             // 1/sqrt(64)
            float c = s1[r] * 0.125f;
            float rowmax = rmax16(fmaxf(a, c));
            float newm = fmaxf(mst[r], rowmax);
            float p0 = __expf(a - newm);
            float p1 = __expf(c - newm);
            float corr = __expf(mst[r] - newm);
            lst[r] = lst[r] * corr + rsum16(p0 + p1);
            mst[r] = newm;
            o0[r] *= corr; o1[r] *= corr; o2[r] *= corr; o3[r] *= corr;
            int rw = r + 8 * hlf;                 // C-layout row
            ldsP[rw * 32 + n]      = f2bf(p0);
            ldsP[rw * 32 + 16 + n] = f2bf(p1);
        }
        __syncthreads();
        v16bf pa = load_a_lds(&ldsP[0]);
        o0 = wmma_bf16(pa, load_bt_tile(vp, SS, 0,  key0), o0);
        o1 = wmma_bf16(pa, load_bt_tile(vp, SS, 16, key0), o1);
        o2 = wmma_bf16(pa, load_bt_tile(vp, SS, 32, key0), o2);
        o3 = wmma_bf16(pa, load_bt_tile(vp, SS, 48, key0), o3);
        __syncthreads();
    }

#pragma unroll
    for (int r = 0; r < 8; ++r) {
        int m = qt * 16 + r + 8 * hlf;
        size_t tok = (size_t)b * SS + m;
        float inv = 1.f / lst[r];
        ctxB[tok * DDIM + h * HDIM + 0 * 16 + n] = f2bf(o0[r] * inv);
        ctxB[tok * DDIM + h * HDIM + 1 * 16 + n] = f2bf(o1[r] * inv);
        ctxB[tok * DDIM + h * HDIM + 2 * 16 + n] = f2bf(o2[r] * inv);
        ctxB[tok * DDIM + h * HDIM + 3 * 16 + n] = f2bf(o3[r] * inv);
    }
}

// ---------------- layernorm over D=512, one block (256 thr) per row ----------------
__global__ void k_layernorm(const float* __restrict__ y,
                            const float* __restrict__ g,
                            const float* __restrict__ be,
                            float* __restrict__ outF,
                            unsigned short* __restrict__ outB) {
    __shared__ float s1[256], s2[256];
    int tid = threadIdx.x;
    size_t row = blockIdx.x;
    const float* yr = y + row * DDIM;
    float a = yr[tid], b = yr[tid + 256];
    s1[tid] = a + b;
    s2[tid] = a * a + b * b;
    __syncthreads();
#pragma unroll
    for (int off = 128; off > 0; off >>= 1) {
        if (tid < off) { s1[tid] += s1[tid + off]; s2[tid] += s2[tid + off]; }
        __syncthreads();
    }
    float mu = s1[0] * (1.f / DDIM);
    float var = s2[0] * (1.f / DDIM) - mu * mu;
    float inv = rsqrtf(var + 1e-5f);
#pragma unroll
    for (int t = 0; t < 2; ++t) {
        int j = tid + t * 256;
        float v = (yr[j] - mu) * inv * g[j] + be[j];
        outF[row * DDIM + j] = v;
        if (outB) outB[row * DDIM + j] = f2bf(v);
    }
}

extern "C" void kernel_launch(void* const* d_in, const int* in_sizes, int n_in,
                              void* d_out, int out_size, void* d_ws, size_t ws_size,
                              hipStream_t stream) {
    (void)in_sizes; (void)n_in; (void)out_size; (void)ws_size;
    const float* src        = (const float*)d_in[0];
    const float* in_proj_w  = (const float*)d_in[1];
    const float* in_proj_b  = (const float*)d_in[2];
    const float* out_proj_w = (const float*)d_in[3];
    const float* out_proj_b = (const float*)d_in[4];
    const float* w1  = (const float*)d_in[5];
    const float* b1  = (const float*)d_in[6];
    const float* w2  = (const float*)d_in[7];
    const float* b2  = (const float*)d_in[8];
    const float* g1  = (const float*)d_in[9];
    const float* be1 = (const float*)d_in[10];
    const float* g2  = (const float*)d_in[11];
    const float* be2 = (const float*)d_in[12];
    float* out = (float*)d_out;

    char* ws = (char*)d_ws;
    size_t off = 0;
    auto alloc = [&](size_t bytes) -> void* {
        void* p = ws + off;
        off += (bytes + 255) & ~(size_t)255;
        return p;
    };
    unsigned short* srcB  = (unsigned short*)alloc((size_t)NTOK * DDIM * 2);
    unsigned short* wqkvB = (unsigned short*)alloc((size_t)3 * DDIM * DDIM * 2);
    unsigned short* woB   = (unsigned short*)alloc((size_t)DDIM * DDIM * 2);
    unsigned short* w1B   = (unsigned short*)alloc((size_t)FFD * DDIM * 2);
    unsigned short* w2B   = (unsigned short*)alloc((size_t)DDIM * FFD * 2);
    unsigned short* qB    = (unsigned short*)alloc((size_t)NTOK * DDIM * 2);
    unsigned short* kB    = (unsigned short*)alloc((size_t)NTOK * DDIM * 2);
    unsigned short* vT    = (unsigned short*)alloc((size_t)NTOK * DDIM * 2);
    unsigned short* ctxB  = (unsigned short*)alloc((size_t)NTOK * DDIM * 2);
    float*          y1    = (float*)alloc((size_t)NTOK * DDIM * 4);
    float*          x     = (float*)alloc((size_t)NTOK * DDIM * 4);
    unsigned short* xB    = (unsigned short*)alloc((size_t)NTOK * DDIM * 2);
    unsigned short* hB    = (unsigned short*)alloc((size_t)NTOK * FFD * 2);
    float*          y2    = (float*)alloc((size_t)NTOK * DDIM * 4);

    auto cvt = [&](const float* p, unsigned short* q, int n) {
        k_f32_to_bf16<<<(n + 255) / 256, 256, 0, stream>>>(p, q, n);
    };
    cvt(src,        srcB,  NTOK * DDIM);
    cvt(in_proj_w,  wqkvB, 3 * DDIM * DDIM);
    cvt(out_proj_w, woB,   DDIM * DDIM);
    cvt(w1,         w1B,   FFD * DDIM);
    cvt(w2,         w2B,   DDIM * FFD);

    // QKV projection + head scatter: N=1536, workgroup covers 16x256
    k_gemm_async<4, 1><<<dim3(3 * DDIM / 256, NTOK / 16), 128, 0, stream>>>(
        srcB, wqkvB, in_proj_b, nullptr, nullptr, nullptr,
        qB, kB, vT, 3 * DDIM, DDIM, 0);

    // attention
    k_attn<<<dim3(SS / 16, HH, BB), 32, 0, stream>>>(qB, kB, vT, ctxB);

    // out_proj + residual(src) -> y1
    k_gemm_async<4, 0><<<dim3(DDIM / 256, NTOK / 16), 128, 0, stream>>>(
        ctxB, woB, out_proj_b, src, y1, nullptr,
        nullptr, nullptr, nullptr, DDIM, DDIM, 0);

    // LN1 -> x (f32) + xB (bf16)
    k_layernorm<<<NTOK, 256, 0, stream>>>(y1, g1, be1, x, xB);

    // FFN layer 1: relu(x @ w1^T + b1) -> hB (bf16); N=128, TN=2 (4 waves x 32 cols)
    k_gemm_async<2, 0><<<dim3(FFD / 128, NTOK / 16), 128, 0, stream>>>(
        xB, w1B, b1, nullptr, nullptr, hB,
        nullptr, nullptr, nullptr, FFD, DDIM, 1);

    // FFN layer 2: h @ w2^T + b2 + x -> y2 ; K=128
    k_gemm_async<4, 0><<<dim3(DDIM / 256, NTOK / 16), 128, 0, stream>>>(
        hB, w2B, b2, x, y2, nullptr,
        nullptr, nullptr, nullptr, DDIM, FFD, 0);

    // LN2 -> d_out
    k_layernorm<<<NTOK, 256, 0, stream>>>(y2, g2, be2, out, nullptr);
}
